// QuantumMLP_12524124635732
// MI455X (gfx1250) — matrix operand order repaired and checked by reading, not compiled, not run on hardware
//
#include <hip/hip_runtime.h>

#define B_SZ   1024
#define IN_SZ  1024
#define OUT_SZ 1024
#define NQ     10
#define G_SZ   4
#define DEPTH  4
#define QDIM   1024

typedef __attribute__((ext_vector_type(16))) __bf16 v16bf;
typedef __attribute__((ext_vector_type(8)))  __bf16 v8bf;
typedef __attribute__((ext_vector_type(8)))  float  v8f;

// ---------------------------------------------------------------------------
// x * inp_scale -> f32 (for reup) and bf16 (for WMMA GEMM1)
// ---------------------------------------------------------------------------
__global__ __launch_bounds__(256) void k_scale_convert_x(
    const float* __restrict__ x, const float* __restrict__ scl,
    float* __restrict__ xs, __bf16* __restrict__ xb, int n) {
  int i = blockIdx.x * 256 + threadIdx.x;
  if (i < n) {
    float v = x[i] * scl[i & (IN_SZ - 1)];
    xs[i] = v;
    xb[i] = (__bf16)v;
  }
}

// ---------------------------------------------------------------------------
// Weight convert + transpose: src (N,K) f32 row-major -> dst (K,N) bf16.
// 32x32 LDS tile, coalesced read and write.
// ---------------------------------------------------------------------------
__global__ __launch_bounds__(256) void k_transpose_bf16(
    const float* __restrict__ src, __bf16* __restrict__ dst, int N, int K,
    int tilesK) {
  __shared__ __bf16 tile[32][33];
  const int tx = threadIdx.x & 31;
  const int ty = threadIdx.x >> 5;           // 0..7
  const int tn = blockIdx.x / tilesK;
  const int tk = blockIdx.x - tn * tilesK;
  const int n0 = tn * 32, k0 = tk * 32;
#pragma unroll
  for (int j = 0; j < 32; j += 8)
    tile[ty + j][tx] = (__bf16)src[(size_t)(n0 + ty + j) * K + k0 + tx];
  __syncthreads();
#pragma unroll
  for (int j = 0; j < 32; j += 8)
    dst[(size_t)(k0 + ty + j) * N + n0 + tx] = tile[tx][ty + j];
}

// ---------------------------------------------------------------------------
// WMMA GEMM: C(M,N) = A(M,K) * Wt(K,N) + bias ; A,Wt bf16, C f32.
// One wave -> 32x32 tile (2x2 WMMA). A fragments loaded as b128 vectors:
//  A: lane's 16 elems = two contiguous 8-elem runs (ISA A 16x32 layout)
//  B: lane = N column of Wt, elems = K in same order as A (strided gather)
// ---------------------------------------------------------------------------
__global__ __launch_bounds__(256) void k_wmma_gemm(
    const __bf16* __restrict__ A, const __bf16* __restrict__ Wt,
    const float* __restrict__ bias, float* __restrict__ C,
    int M, int N, int K, int tilesN) {
  const int wave = (blockIdx.x * 256 + threadIdx.x) >> 5;
  const int lane = threadIdx.x & 31;
  const int tm = wave / tilesN;
  const int tn = wave - tm * tilesN;
  const int m0 = tm * 32;
  const int n0 = tn * 32;
  const int mrow = lane & 15;
  const int half = lane >> 4;

  v8f acc00 = {}, acc01 = {}, acc10 = {}, acc11 = {};

  const __bf16* a0p = A + (size_t)(m0 + mrow) * K + half * 8;
  const __bf16* a1p = a0p + (size_t)16 * K;
  const __bf16* wp  = Wt + (size_t)(half * 8) * N + n0 + mrow;

  for (int k0 = 0; k0 < K; k0 += 32) {
    const v8bf a0lo = *(const v8bf*)(a0p + k0);
    const v8bf a0hi = *(const v8bf*)(a0p + k0 + 16);
    const v8bf a1lo = *(const v8bf*)(a1p + k0);
    const v8bf a1hi = *(const v8bf*)(a1p + k0 + 16);
    const __bf16* wk = wp + (size_t)k0 * N;
    v16bf bf0, bf1;
#pragma unroll
    for (int s = 0; s < 8; ++s) {
      bf0[s]     = wk[(size_t)s * N];
      bf0[s + 8] = wk[(size_t)(16 + s) * N];
      bf1[s]     = wk[(size_t)s * N + 16];
      bf1[s + 8] = wk[(size_t)(16 + s) * N + 16];
    }

    const v16bf af0 = __builtin_shufflevector(a0lo, a0hi, 0,1,2,3,4,5,6,7,8,9,10,11,12,13,14,15);
    const v16bf af1 = __builtin_shufflevector(a1lo, a1hi, 0,1,2,3,4,5,6,7,8,9,10,11,12,13,14,15);


    acc00 = __builtin_amdgcn_wmma_f32_16x16x32_bf16(false, af0, false, bf0, (short)0, acc00, false, false);
    acc01 = __builtin_amdgcn_wmma_f32_16x16x32_bf16(false, af0, false, bf1, (short)0, acc01, false, false);
    acc10 = __builtin_amdgcn_wmma_f32_16x16x32_bf16(false, af1, false, bf0, (short)0, acc10, false, false);
    acc11 = __builtin_amdgcn_wmma_f32_16x16x32_bf16(false, af1, false, bf1, (short)0, acc11, false, false);
  }

  const int ncol = lane & 15;
  const int nA = n0 + ncol;
  const int nB = nA + 16;
  const float bA = bias[nA];
  const float bB = bias[nB];
#pragma unroll
  for (int rr = 0; rr < 8; ++rr) {
    const int mA = m0 + rr + half * 8;
    const int mB = mA + 16;
    C[(size_t)mA * N + nA] = acc00[rr] + bA;
    C[(size_t)mA * N + nB] = acc01[rr] + bB;
    C[(size_t)mB * N + nA] = acc10[rr] + bA;
    C[(size_t)mB * N + nB] = acc11[rr] + bB;
  }
}

// ---------------------------------------------------------------------------
// Re-upload angles: one wave per output; coalesced strided loads + shfl reduce
// ang = tanh(xs @ reup_w^T + b) * pi   (B x 40)
// ---------------------------------------------------------------------------
__global__ __launch_bounds__(256) void k_reup(
    const float* __restrict__ xs, const float* __restrict__ w,
    const float* __restrict__ b, float* __restrict__ ang, int n) {
  const int wid  = (blockIdx.x * 256 + threadIdx.x) >> 5;
  const int lane = threadIdx.x & 31;
  if (wid >= n) return;
  const int row = wid / (NQ * G_SZ);
  const int j   = wid - row * (NQ * G_SZ);
  const float* xr = xs + (size_t)row * IN_SZ;
  const float* wr = w  + (size_t)j   * IN_SZ;
  float acc = 0.f;
#pragma unroll 8
  for (int k = lane; k < IN_SZ; k += 32) acc = fmaf(xr[k], wr[k], acc);
#pragma unroll
  for (int m = 16; m >= 1; m >>= 1) acc += __shfl_xor(acc, m, 32);
  if (lane == 0) ang[wid] = tanhf(acc + b[j]) * 3.14159265358979323846f;
}

// ---------------------------------------------------------------------------
// Quantum simulator: one wave32 per state; state register-resident.
// Element e = r*32 + lane; wire w acts on bit (9-w):
//   bits 5..9 -> register-index bits (intra-lane butterflies)
//   bits 0..4 -> lane bits (shfl_xor butterflies)
// ---------------------------------------------------------------------------
__device__ __forceinline__ float lshfl_xor(float v, int m) {
  return __shfl_xor(v, m, 32);
}

template<int BIT>
__device__ __forceinline__ void g_ry(float (&re)[32], float (&im)[32],
                                     float c, float s, int lane) {
  if constexpr (BIT >= 5) {
    constexpr int m = 1 << (BIT - 5);
#pragma unroll
    for (int r = 0; r < 32; ++r) {
      if ((r & m) == 0) {
        const int q = r | m;
        float r0 = re[r], r1 = re[q], i0 = im[r], i1 = im[q];
        re[r] = c * r0 - s * r1;  re[q] = s * r0 + c * r1;
        im[r] = c * i0 - s * i1;  im[q] = s * i0 + c * i1;
      }
    }
  } else {
    constexpr int lm = 1 << BIT;
    const float sg = (lane & lm) ? s : -s;
#pragma unroll
    for (int r = 0; r < 32; ++r) {
      float pr = lshfl_xor(re[r], lm);
      float pi = lshfl_xor(im[r], lm);
      re[r] = c * re[r] + sg * pr;
      im[r] = c * im[r] + sg * pi;
    }
  }
}

template<int BIT>
__device__ __forceinline__ void g_rx(float (&re)[32], float (&im)[32],
                                     float c, float s, int lane) {
  if constexpr (BIT >= 5) {
    constexpr int m = 1 << (BIT - 5);
#pragma unroll
    for (int r = 0; r < 32; ++r) {
      if ((r & m) == 0) {
        const int q = r | m;
        float r0 = re[r], r1 = re[q], i0 = im[r], i1 = im[q];
        re[r] = c * r0 + s * i1;  im[r] = c * i0 - s * r1;
        re[q] = c * r1 + s * i0;  im[q] = c * i1 - s * r0;
      }
    }
  } else {
    constexpr int lm = 1 << BIT;
#pragma unroll
    for (int r = 0; r < 32; ++r) {
      float pr = lshfl_xor(re[r], lm);
      float pi = lshfl_xor(im[r], lm);
      float nr = c * re[r] + s * pi;
      float ni = c * im[r] - s * pr;
      re[r] = nr; im[r] = ni;
    }
  }
}

template<int CBIT, int TBIT>
__device__ __forceinline__ void g_cnot(float (&re)[32], float (&im)[32], int lane) {
  if constexpr (TBIT >= 5) {
    constexpr int tm = 1 << (TBIT - 5);
    if constexpr (CBIT >= 5) {
      constexpr int cm = 1 << (CBIT - 5);
#pragma unroll
      for (int r = 0; r < 32; ++r) {
        if ((r & tm) == 0 && (r & cm) != 0) {
          const int q = r | tm;
          float t = re[r]; re[r] = re[q]; re[q] = t;
          t = im[r]; im[r] = im[q]; im[q] = t;
        }
      }
    } else {
      constexpr int cl = 1 << CBIT;
      const bool cs = (lane & cl) != 0;
#pragma unroll
      for (int r = 0; r < 32; ++r) {
        if ((r & tm) == 0) {
          const int q = r | tm;
          float a = re[r], b = re[q];
          re[r] = cs ? b : a;  re[q] = cs ? a : b;
          a = im[r]; b = im[q];
          im[r] = cs ? b : a;  im[q] = cs ? a : b;
        }
      }
    }
  } else {
    constexpr int tl = 1 << TBIT;
#pragma unroll
    for (int r = 0; r < 32; ++r) {
      float pr = lshfl_xor(re[r], tl);
      float pi = lshfl_xor(im[r], tl);
      bool cs;
      if constexpr (CBIT >= 5) cs = (r & (1 << (CBIT - 5))) != 0;
      else                     cs = (lane & (1 << CBIT)) != 0;
      re[r] = cs ? pr : re[r];
      im[r] = cs ? pi : im[r];
    }
  }
}

template<int BIT>
__device__ __forceinline__ void g_u3(float (&re)[32], float (&im)[32],
                                     float m00r, float m00i, float m01r, float m01i,
                                     float m10r, float m10i, float m11r, float m11i,
                                     int lane) {
  if constexpr (BIT >= 5) {
    constexpr int m = 1 << (BIT - 5);
#pragma unroll
    for (int r = 0; r < 32; ++r) {
      if ((r & m) == 0) {
        const int q = r | m;
        float z0r = re[r], z0i = im[r], z1r = re[q], z1i = im[q];
        re[r] = m00r*z0r - m00i*z0i + m01r*z1r - m01i*z1i;
        im[r] = m00r*z0i + m00i*z0r + m01r*z1i + m01i*z1r;
        re[q] = m10r*z0r - m10i*z0i + m11r*z1r - m11i*z1i;
        im[q] = m10r*z0i + m10i*z0r + m11r*z1i + m11i*z1r;
      }
    }
  } else {
    constexpr int lm = 1 << BIT;
    const bool hi = (lane & lm) != 0;
    const float ar = hi ? m11r : m00r, ai = hi ? m11i : m00i;
    const float br = hi ? m10r : m01r, bi = hi ? m10i : m01i;
#pragma unroll
    for (int r = 0; r < 32; ++r) {
      float pr = lshfl_xor(re[r], lm);
      float pi = lshfl_xor(im[r], lm);
      float nr = ar*re[r] - ai*im[r] + br*pr - bi*pi;
      float ni = ar*im[r] + ai*re[r] + br*pi + bi*pr;
      re[r] = nr; im[r] = ni;
    }
  }
}

__global__ __launch_bounds__(256) void k_qsim(
    const float* __restrict__ xq, const float* __restrict__ ang,
    const float* __restrict__ qw, const float* __restrict__ us,
    const float* __restrict__ mw, __bf16* __restrict__ probs) {
  const int wid  = (blockIdx.x * 256 + threadIdx.x) >> 5;  // state s = b*G + g
  const int lane = threadIdx.x & 31;
  const int b = wid >> 2;
  const int g = wid & 3;

  const float* row = xq + (size_t)wid * QDIM;
  float re[32], im[32];
  float ss = 0.f;
#pragma unroll
  for (int r = 0; r < 32; ++r) {
    float v = row[r * 32 + lane];
    re[r] = v; im[r] = 0.f;
    ss = fmaf(v, v, ss);
  }
#pragma unroll
  for (int m = 16; m >= 1; m >>= 1) ss += lshfl_xor(ss, m);
  const float inv = 1.f / (sqrtf(ss) + 1e-9f);
#pragma unroll
  for (int r = 0; r < 32; ++r) re[r] *= inv;

  float th[NQ];
  const float* arp = ang + (size_t)b * (NQ * G_SZ) + g * NQ;
#pragma unroll
  for (int i = 0; i < NQ; ++i) th[i] = arp[i];

  for (int l = 0; l < DEPTH; ++l) {
    const float* usl = us + ((size_t)g * DEPTH + l) * NQ;
    const float* qwl = qw + (((size_t)g * DEPTH + l) * NQ) * 3;

#define RY_ANG(i) { float t = 0.5f * th[i] * usl[i]; g_ry<9-(i)>(re, im, cosf(t), sinf(t), lane); }
    RY_ANG(0) RY_ANG(1) RY_ANG(2) RY_ANG(3) RY_ANG(4)
    RY_ANG(5) RY_ANG(6) RY_ANG(7) RY_ANG(8) RY_ANG(9)
#undef RY_ANG

#define RXRY(i) { float t0 = 0.5f * qwl[(i)*3+0]; g_rx<9-(i)>(re, im, cosf(t0), sinf(t0), lane); \
                  float t1 = 0.5f * qwl[(i)*3+1]; g_ry<9-(i)>(re, im, cosf(t1), sinf(t1), lane); }
    RXRY(0) RXRY(1) RXRY(2) RXRY(3) RXRY(4)
    RXRY(5) RXRY(6) RXRY(7) RXRY(8) RXRY(9)
#undef RXRY

    g_cnot<9,8>(re, im, lane);
    g_cnot<8,7>(re, im, lane);
    g_cnot<7,6>(re, im, lane);
    g_cnot<6,5>(re, im, lane);
    g_cnot<5,4>(re, im, lane);
    g_cnot<4,3>(re, im, lane);
    g_cnot<3,2>(re, im, lane);
    g_cnot<2,1>(re, im, lane);
    g_cnot<1,0>(re, im, lane);
    g_cnot<0,9>(re, im, lane);

#define RY2(i) { float t = 0.5f * qwl[(i)*3+2]; g_ry<9-(i)>(re, im, cosf(t), sinf(t), lane); }
    RY2(0) RY2(1) RY2(2) RY2(3) RY2(4)
    RY2(5) RY2(6) RY2(7) RY2(8) RY2(9)
#undef RY2
  }

  const float* mwg = mw + (size_t)g * NQ * 3;
#define U3G(i) { float t = 0.5f * mwg[(i)*3+0]; float ph = mwg[(i)*3+1]; float la = mwg[(i)*3+2]; \
                 float c = cosf(t), s = sinf(t); \
                 float clm = cosf(la), slm = sinf(la); \
                 float cph = cosf(ph), sph = sinf(ph); \
                 float cpl = cosf(ph + la), spl = sinf(ph + la); \
                 g_u3<9-(i)>(re, im, c, 0.f, -clm*s, -slm*s, cph*s, sph*s, cpl*c, spl*c, lane); }
  U3G(0) U3G(1) U3G(2) U3G(3) U3G(4)
  U3G(5) U3G(6) U3G(7) U3G(8) U3G(9)
#undef U3G

  __bf16* pr = probs + (size_t)wid * QDIM;
#pragma unroll
  for (int r = 0; r < 32; ++r) {
    float p = re[r]*re[r] + im[r]*im[r];
    pr[r*32 + lane] = (__bf16)p;
  }
}

// ---------------------------------------------------------------------------
// Launch
// ---------------------------------------------------------------------------
extern "C" void kernel_launch(void* const* d_in, const int* in_sizes, int n_in,
                              void* d_out, int out_size, void* d_ws, size_t ws_size,
                              hipStream_t stream) {
  const float* x       = (const float*)d_in[0];
  const float* inp_scl = (const float*)d_in[1];
  const float* piw     = (const float*)d_in[2];
  const float* pib     = (const float*)d_in[3];
  const float* ruw     = (const float*)d_in[4];
  const float* rub     = (const float*)d_in[5];
  const float* qwts    = (const float*)d_in[6];
  const float* uscl    = (const float*)d_in[7];
  const float* mw      = (const float*)d_in[8];
  const float* pow_    = (const float*)d_in[9];
  const float* pob     = (const float*)d_in[10];
  float* out = (float*)d_out;

  char* ws = (char*)d_ws;
  float*  xs  = (float*)(ws);                          // 4 MiB  (x * scale, f32)
  __bf16* xb  = (__bf16*)(ws + ((size_t)4  << 20));    // 2 MiB  (x * scale, bf16)
  __bf16* w1t = (__bf16*)(ws + ((size_t)6  << 20));    // 8 MiB  (proj_in_w^T bf16, (K=1024, N=4096))
  __bf16* w2t = (__bf16*)(ws + ((size_t)14 << 20));    // 8 MiB  (proj_out_w^T bf16, (K=4096, N=1024))
  float*  xq  = (float*)(ws + ((size_t)22 << 20));     // 16 MiB (GEMM1 out f32)
  float*  ang = (float*)(ws + ((size_t)38 << 20));     // 160 KiB (angles)
  __bf16* prb = (__bf16*)(ws + ((size_t)39 << 20));    // 8 MiB  (probs bf16)

  k_scale_convert_x<<<(B_SZ * IN_SZ) / 256, 256, 0, stream>>>(x, inp_scl, xs, xb, B_SZ * IN_SZ);

  {  // proj_in_w (4096,1024) -> w1t (1024,4096)
    const int Nw = G_SZ * QDIM, Kw = IN_SZ, tilesK = Kw / 32;
    k_transpose_bf16<<<(Nw / 32) * tilesK, 256, 0, stream>>>(piw, w1t, Nw, Kw, tilesK);
  }
  {  // proj_out_w (1024,4096) -> w2t (4096,1024)
    const int Nw = OUT_SZ, Kw = G_SZ * QDIM, tilesK = Kw / 32;
    k_transpose_bf16<<<(Nw / 32) * tilesK, 256, 0, stream>>>(pow_, w2t, Nw, Kw, tilesK);
  }

  {  // GEMM1: xq = xb @ piw^T + pib   (M=1024, N=4096, K=1024)
    const int M = B_SZ, N = G_SZ * QDIM, K = IN_SZ;
    const int tilesN = N / 32;
    const int waves = (M / 32) * (N / 32);
    k_wmma_gemm<<<waves / 8, 256, 0, stream>>>(xb, w1t, pib, xq, M, N, K, tilesN);
  }

  k_reup<<<(B_SZ * NQ * G_SZ) / 8, 256, 0, stream>>>(xs, ruw, rub, ang, B_SZ * NQ * G_SZ);

  k_qsim<<<(B_SZ * G_SZ) / 8, 256, 0, stream>>>(xq, ang, qwts, uscl, mw, prb);

  {  // GEMM3: out = probs @ pow^T + pob   (M=1024, N=1024, K=4096)
    const int M = B_SZ, N = OUT_SZ, K = G_SZ * QDIM;
    const int tilesN = N / 32;
    const int waves = (M / 32) * (N / 32);
    k_wmma_gemm<<<waves / 8, 256, 0, stream>>>(prb, w2t, pob, out, M, N, K, tilesN);
  }
}
